// PolarRnn_72464688218466
// MI455X (gfx1250) — matrix operand order repaired and checked by reading, not compiled, not run on hardware
//
#include <hip/hip_runtime.h>
#include <hip/hip_bf16.h>
#include <stdint.h>

// Problem dims (fixed by the reference)
#define BATCH  4
#define SEQ    2048
#define DIM    1024
#define NHEAD  16
#define HDIM   64
#define MROWS  (BATCH * SEQ)   // 8192 tokens

// ---------------------------------------------------------------------------
// Types for CDNA5 WMMA (wave32): v_wmma_f32_16x16x32_bf16
// ---------------------------------------------------------------------------
typedef __attribute__((ext_vector_type(16))) __bf16          v16bf;
typedef __attribute__((ext_vector_type(8)))  float           v8f;
typedef __attribute__((ext_vector_type(8)))  unsigned short  us8;

__device__ __forceinline__ unsigned short f2bf_bits(float f) {
  unsigned u = __float_as_uint(f);
  unsigned r = u + 0x7FFFu + ((u >> 16) & 1u);   // round-to-nearest-even
  return (unsigned short)(r >> 16);
}

// ---------------------------------------------------------------------------
// f32 -> bf16 flat convert
// ---------------------------------------------------------------------------
__global__ __launch_bounds__(256) void cvt_bf16_kernel(
    const float* __restrict__ in, unsigned short* __restrict__ out, size_t n)
{
  size_t i = (size_t)blockIdx.x * blockDim.x + threadIdx.x;
  size_t stride = (size_t)gridDim.x * blockDim.x;
  for (; i < n; i += stride) out[i] = f2bf_bits(in[i]);
}

// f32 [K,N] -> bf16 [N,K] (transpose so GEMM B-fragments are contiguous in K)
__global__ __launch_bounds__(256) void cvt_bf16_T_kernel(
    const float* __restrict__ in, unsigned short* __restrict__ out, int K, int N)
{
  size_t i = (size_t)blockIdx.x * blockDim.x + threadIdx.x;
  size_t total = (size_t)K * N;
  size_t stride = (size_t)gridDim.x * blockDim.x;
  for (; i < total; i += stride) {
    int k = (int)(i / (size_t)N);
    int n = (int)(i - (size_t)k * N);
    out[(size_t)n * K + k] = f2bf_bits(in[i]);
  }
}

// ---------------------------------------------------------------------------
// WMMA GEMM: C[M,N] = act(A[M,K] @ B[K,N]),  A bf16 row-major, B given
// transposed bf16 [N,K].  Register-blocked 16x(16*NT) strip per wave with NT
// accumulators; one A fragment reused across NT B fragments per K-step.
// Explicitly software-pipelined 2 deep over K (K % 64 == 0): fragment set for
// step i+1 is loaded while the NT WMMAs of step i issue, so waits become
// partial s_wait_loadcnt instead of full drains.  Prefetches are speculative
// and unconditional (translation faults silently dropped per ISA 10.5).
// Fragment layouts used here (A and B share one contiguous k order):
//   A 16x32 bf16:  lane<16 -> K[0..15] of row M=lane
//                  lane>=16 -> K[16..31] of row M=lane-16
//   B 32x16 bf16:  lane<16 -> K[0..15]  of col N=lane
//                  lane>=16 -> K[16..31] of col N=lane-16
//   C f32 16x16:   VGPR r: lane<16 -> (M=r, N=lane); lane>=16 -> (M=8+r, N=lane-16)
// ACT: 0=none 1=silu 2=sigmoid 3=logsigmoid 4=-sigmoid
// ---------------------------------------------------------------------------
template<int ACT, int OUT_BF, int NT>
__global__ __launch_bounds__(256) void wmma_gemm_kernel(
    const unsigned short* __restrict__ A,
    const unsigned short* __restrict__ Bt,
    void* __restrict__ Cout,
    int M, int N, int K)
{
  const int wave = threadIdx.x >> 5;
  const int lane = threadIdx.x & 31;
  const int tiles_n = N / (16 * NT);
  const int tiles_m = M >> 4;
  const int tile = blockIdx.x * 8 + wave;
  if (tile >= tiles_m * tiles_n) return;            // wave-uniform: EXEC stays full
  const int tm = tile / tiles_n;
  const int tn = tile - tm * tiles_n;
  const int half = lane >> 4;                       // 0 or 1
  const int l16  = lane & 15;

  const unsigned short* arow = A + (size_t)(tm * 16 + l16) * K + half * 16;
  const unsigned short* brow[NT];
  #pragma unroll
  for (int t = 0; t < NT; ++t)
    brow[t] = Bt + (size_t)((tn * NT + t) * 16 + l16) * K + half * 16;

  v8f acc[NT];
  #pragma unroll
  for (int t = 0; t < NT; ++t)
    acc[t] = (v8f){0.f, 0.f, 0.f, 0.f, 0.f, 0.f, 0.f, 0.f};

  union Frag { v16bf v; us8 h[2]; };
  Frag a0, a1, b0[NT], b1[NT];

  auto ldA = [&](Frag& f, int k) {
    f.h[0] = *(const us8*)(arow + k);               // K[kk0 .. kk0+7]
    f.h[1] = *(const us8*)(arow + k + 8);           // K[kk0+8 .. kk0+15]
  };
  auto ldB = [&](Frag* f, int k) {
    #pragma unroll
    for (int t = 0; t < NT; ++t) {
      f[t].h[0] = *(const us8*)(brow[t] + k);       // K[kk0 .. kk0+7]
      f[t].h[1] = *(const us8*)(brow[t] + k + 8);   // K[kk0+8 .. kk0+15]
    }
  };
  auto mm = [&](Frag& a, Frag* b) {
    #pragma unroll
    for (int t = 0; t < NT; ++t)
      acc[t] = __builtin_amdgcn_wmma_f32_16x16x32_bf16(
          false, a.v, false, b[t].v, (short)0, acc[t], false, false);
  };

  // prologue: fragment set for K-step 0 in flight
  ldA(a0, 0); ldB(b0, 0);

  int k0 = 0;
  for (; k0 + 64 < K; k0 += 64) {                   // steady state: K%64==0
    ldA(a1, k0 + 32); ldB(b1, k0 + 32);             // next set in flight
    __builtin_prefetch(arow + k0 + 64, 0, 1);       // speculative, branchless
    #pragma unroll
    for (int t = 0; t < NT; ++t)
      __builtin_prefetch(brow[t] + k0 + 64, 0, 1);
    mm(a0, b0);                                     // consume set 0
    ldA(a0, k0 + 64); ldB(b0, k0 + 64);             // refill set 0
    mm(a1, b1);                                     // consume set 1
  }
  // epilogue: k0 == K-64, set0 already loaded
  ldA(a1, k0 + 32); ldB(b1, k0 + 32);
  mm(a0, b0);
  mm(a1, b1);

  const int mbase = tm * 16 + half * 8;
  #pragma unroll
  for (int t = 0; t < NT; ++t) {
    const int n = (tn * NT + t) * 16 + l16;
    #pragma unroll
    for (int r = 0; r < 8; ++r) {
      float x = acc[t][r];
      if (ACT == 1)      x = x / (1.f + __expf(-x));                 // silu
      else if (ACT == 2) x = 1.f / (1.f + __expf(-x));               // sigmoid
      else if (ACT == 3) { float mn = fminf(x, 0.f);                 // log_sigmoid
                           x = mn - __logf(1.f + __expf(-fabsf(x))); }
      else if (ACT == 4) x = -1.f / (1.f + __expf(-x));              // -sigmoid
      size_t idx = (size_t)(mbase + r) * N + n;
      if (OUT_BF) ((unsigned short*)Cout)[idx] = f2bf_bits(x);
      else        ((float*)Cout)[idx]          = x;
    }
  }
}

// ---------------------------------------------------------------------------
// Per-(token,head) prep: l2-normalize k over HD, egk=exp(gk),
// a = k_norm * gamma_h * egk.  One wave32 per (token,head), 2 elems/lane.
// k and gk are updated in place (k -> k_norm, gk -> exp(gk)).
// ---------------------------------------------------------------------------
__global__ __launch_bounds__(256) void prep_kernel(
    float* __restrict__ k, float* __restrict__ gk,
    const float* __restrict__ gamma, float* __restrict__ a_out)
{
  const int wid  = blockIdx.x * 8 + (threadIdx.x >> 5);
  const int lane = threadIdx.x & 31;
  const int tok  = wid / NHEAD;
  const int h    = wid - tok * NHEAD;
  const size_t base = (size_t)tok * DIM + h * HDIM;

  float k0 = k[base + lane], k1 = k[base + lane + 32];
  float ss = k0 * k0 + k1 * k1;
  #pragma unroll
  for (int off = 16; off >= 1; off >>= 1) ss += __shfl_xor(ss, off, 32);
  const float inv = 1.f / fmaxf(sqrtf(ss), 1e-12f);

  const float g = gamma[(size_t)tok * NHEAD + h];   // already -sigmoid(...)
  float e0 = __expf(gk[base + lane]);
  float e1 = __expf(gk[base + lane + 32]);
  float kn0 = k0 * inv, kn1 = k1 * inv;

  k[base + lane]        = kn0;
  k[base + lane + 32]   = kn1;
  gk[base + lane]       = e0;
  gk[base + lane + 32]  = e1;
  a_out[base + lane]      = kn0 * g * e0;
  a_out[base + lane + 32] = kn1 * g * e1;
}

// ---------------------------------------------------------------------------
// DPLR delta-rule scan.  State columns are independent: lane j owns S[:,j]
// (64 f32, fully register-resident via full unroll).  One block (2 waves) per
// (b,h).  Per-step broadcasts (q,k,a,egk over the key dim) staged in LDS for
// RCHUNK steps at a time (same-address reads -> bank-conflict-free broadcast).
//   S = diag(egk) S;  sab_j = a . S[:,j];  S[:,j] += k*(sab_j + v_j);
//   o_j = q . S[:,j]
// ---------------------------------------------------------------------------
#define RCHUNK 8
__global__ __launch_bounds__(64) void recurrence_kernel(
    const float* __restrict__ q, const float* __restrict__ k,
    const float* __restrict__ v, const float* __restrict__ a,
    const float* __restrict__ egk, float* __restrict__ o)
{
  const int bh = blockIdx.x;
  const int b = bh / NHEAD, h = bh - b * NHEAD;
  const int j = threadIdx.x;            // state column (v index), 0..63

  __shared__ float sq[RCHUNK][HDIM], sk[RCHUNK][HDIM],
                   sa[RCHUNK][HDIM], se[RCHUNK][HDIM];

  float S[HDIM];
  #pragma unroll
  for (int i = 0; i < HDIM; ++i) S[i] = 0.f;

  const size_t tok0 = (size_t)b * SEQ;

  for (int t0 = 0; t0 < SEQ; t0 += RCHUNK) {
    for (int i = threadIdx.x; i < RCHUNK * HDIM; i += 64) {
      int tc = i >> 6, e = i & 63;
      size_t off = (tok0 + t0 + tc) * DIM + h * HDIM + e;
      sq[tc][e] = q[off];  sk[tc][e] = k[off];
      sa[tc][e] = a[off];  se[tc][e] = egk[off];
    }
    __syncthreads();

    #pragma unroll 1
    for (int tc = 0; tc < RCHUNK; ++tc) {
      size_t off = (tok0 + t0 + tc) * DIM + h * HDIM + j;
      float vj  = v[off];
      float sab = 0.f;
      #pragma unroll
      for (int kk = 0; kk < HDIM; ++kk) {     // decay then a^T S (decayed)
        float s = S[kk] * se[tc][kk];
        S[kk] = s;
        sab += sa[tc][kk] * s;
      }
      float add = sab + vj;
      float oj  = 0.f;
      #pragma unroll
      for (int kk = 0; kk < HDIM; ++kk) {     // rank-1 update then q^T S
        float s = S[kk] + sk[tc][kk] * add;
        S[kk] = s;
        oj += sq[tc][kk] * s;
      }
      o[off] = oj;
    }
    __syncthreads();
  }
}

// ---------------------------------------------------------------------------
// Output gate + LayerNorm (eps=1e-5, weight only), writes bf16 for final GEMM.
// One block per token; 4 elems/thread; wave32 shuffle + LDS cross-wave reduce.
// ---------------------------------------------------------------------------
__global__ __launch_bounds__(256) void gate_ln_kernel(
    const float* __restrict__ o, const float* __restrict__ og,
    const float* __restrict__ w, unsigned short* __restrict__ out)
{
  const int tok = blockIdx.x;
  const size_t base = (size_t)tok * DIM;
  __shared__ float red[8];

  float vals[4];
  float s = 0.f;
  #pragma unroll
  for (int i = 0; i < 4; ++i) {
    int e = threadIdx.x * 4 + i;
    float x = o[base + e] * og[base + e];
    vals[i] = x; s += x;
  }
  #pragma unroll
  for (int off = 16; off >= 1; off >>= 1) s += __shfl_xor(s, off, 32);
  if ((threadIdx.x & 31) == 0) red[threadIdx.x >> 5] = s;
  __syncthreads();
  float tot = 0.f;
  #pragma unroll
  for (int i = 0; i < 8; ++i) tot += red[i];
  const float mean = tot * (1.f / DIM);
  __syncthreads();

  float vs = 0.f;
  #pragma unroll
  for (int i = 0; i < 4; ++i) { float d = vals[i] - mean; vs += d * d; }
  #pragma unroll
  for (int off = 16; off >= 1; off >>= 1) vs += __shfl_xor(vs, off, 32);
  if ((threadIdx.x & 31) == 0) red[threadIdx.x >> 5] = vs;
  __syncthreads();
  float vtot = 0.f;
  #pragma unroll
  for (int i = 0; i < 8; ++i) vtot += red[i];
  const float rstd = rsqrtf(vtot * (1.f / DIM) + 1e-5f);

  #pragma unroll
  for (int i = 0; i < 4; ++i) {
    int e = threadIdx.x * 4 + i;
    out[base + e] = f2bf_bits((vals[i] - mean) * rstd * w[e]);
  }
}

// ---------------------------------------------------------------------------
// Host orchestration
// ---------------------------------------------------------------------------
extern "C" void kernel_launch(void* const* d_in, const int* in_sizes, int n_in,
                              void* d_out, int out_size, void* d_ws, size_t ws_size,
                              hipStream_t stream)
{
  (void)in_sizes; (void)n_in; (void)out_size; (void)ws_size;

  const float* x      = (const float*)d_in[0];
  const float* Wq     = (const float*)d_in[1];
  const float* Wk     = (const float*)d_in[2];
  const float* Wv     = (const float*)d_in[3];
  const float* Wgamma = (const float*)d_in[4];
  const float* Wf1    = (const float*)d_in[5];
  const float* Wf2    = (const float*)d_in[6];
  const float* Wog1   = (const float*)d_in[7];
  const float* Wog2   = (const float*)d_in[8];
  const float* normw  = (const float*)d_in[9];
  const float* Wo     = (const float*)d_in[10];
  float* y = (float*)d_out;

  char* ws = (char*)d_ws;
  size_t cur = 0;
  auto alloc = [&](size_t bytes) -> void* {
    cur = (cur + 255) & ~(size_t)255;
    void* p = ws + cur;
    cur += bytes;
    return p;
  };

  const size_t MD = (size_t)MROWS * DIM;

  unsigned short* x_bf    = (unsigned short*)alloc(MD * 2);
  unsigned short* Wq_t    = (unsigned short*)alloc((size_t)DIM * DIM * 2);
  unsigned short* Wk_t    = (unsigned short*)alloc((size_t)DIM * DIM * 2);
  unsigned short* Wv_t    = (unsigned short*)alloc((size_t)DIM * DIM * 2);
  unsigned short* Wo_t    = (unsigned short*)alloc((size_t)DIM * DIM * 2);
  unsigned short* Wgam_t  = (unsigned short*)alloc((size_t)DIM * NHEAD * 2);
  unsigned short* Wf1_t   = (unsigned short*)alloc((size_t)DIM * HDIM * 2);
  unsigned short* Wog1_t  = (unsigned short*)alloc((size_t)DIM * HDIM * 2);
  unsigned short* Wf2_t   = (unsigned short*)alloc((size_t)HDIM * DIM * 2);
  unsigned short* Wog2_t  = (unsigned short*)alloc((size_t)HDIM * DIM * 2);

  float* qbuf   = (float*)alloc(MD * 4);
  float* kbuf   = (float*)alloc(MD * 4);   // silu(k) then in-place l2-normed
  float* vbuf   = (float*)alloc(MD * 4);
  float* gkbuf  = (float*)alloc(MD * 4);   // log_sigmoid then in-place exp()
  float* abuf   = (float*)alloc(MD * 4);
  float* ogbuf  = (float*)alloc(MD * 4);
  float* obuf   = (float*)alloc(MD * 4);
  float* gambuf = (float*)alloc((size_t)MROWS * NHEAD * 4);
  unsigned short* f1_bf  = (unsigned short*)alloc((size_t)MROWS * HDIM * 2);
  unsigned short* og1_bf = (unsigned short*)alloc((size_t)MROWS * HDIM * 2);
  unsigned short* oln_bf = (unsigned short*)alloc(MD * 2);

  // ---- stage 0: bf16 conversions (x flat, weights transposed [N,K]) ----
  cvt_bf16_kernel<<<4096, 256, 0, stream>>>(x, x_bf, MD);
  cvt_bf16_T_kernel<<<4096, 256, 0, stream>>>(Wq,     Wq_t,   DIM,  DIM);
  cvt_bf16_T_kernel<<<4096, 256, 0, stream>>>(Wk,     Wk_t,   DIM,  DIM);
  cvt_bf16_T_kernel<<<4096, 256, 0, stream>>>(Wv,     Wv_t,   DIM,  DIM);
  cvt_bf16_T_kernel<<<4096, 256, 0, stream>>>(Wo,     Wo_t,   DIM,  DIM);
  cvt_bf16_T_kernel<<<64,   256, 0, stream>>>(Wgamma, Wgam_t, DIM,  NHEAD);
  cvt_bf16_T_kernel<<<256,  256, 0, stream>>>(Wf1,    Wf1_t,  DIM,  HDIM);
  cvt_bf16_T_kernel<<<256,  256, 0, stream>>>(Wog1,   Wog1_t, DIM,  HDIM);
  cvt_bf16_T_kernel<<<256,  256, 0, stream>>>(Wf2,    Wf2_t,  HDIM, DIM);
  cvt_bf16_T_kernel<<<256,  256, 0, stream>>>(Wog2,   Wog2_t, HDIM, DIM);

  // wave-strip (16 x 16*NT) grids, 8 waves per block
  const int gDD  = (MROWS / 16) * (DIM  / 64) / 8;   // NT=4 -> 1024 blocks
  const int gD64 = (MROWS / 16) * (HDIM / 64) / 8;   // NT=4 -> 64 blocks
  const int gD16 = (MROWS / 16) * (NHEAD / 16) / 8;  // NT=1 -> 64 blocks

  // ---- stage 1: first-level projections (fused activations) ----
  wmma_gemm_kernel<1, 0, 4><<<gDD,  256, 0, stream>>>(x_bf, Wq_t,   qbuf,   MROWS, DIM,   DIM);  // silu
  wmma_gemm_kernel<1, 0, 4><<<gDD,  256, 0, stream>>>(x_bf, Wk_t,   kbuf,   MROWS, DIM,   DIM);  // silu
  wmma_gemm_kernel<0, 0, 4><<<gDD,  256, 0, stream>>>(x_bf, Wv_t,   vbuf,   MROWS, DIM,   DIM);
  wmma_gemm_kernel<4, 0, 1><<<gD16, 256, 0, stream>>>(x_bf, Wgam_t, gambuf, MROWS, NHEAD, DIM);  // -sigmoid
  wmma_gemm_kernel<0, 1, 4><<<gD64, 256, 0, stream>>>(x_bf, Wf1_t,  f1_bf,  MROWS, HDIM,  DIM);  // bf16 out
  wmma_gemm_kernel<0, 1, 4><<<gD64, 256, 0, stream>>>(x_bf, Wog1_t, og1_bf, MROWS, HDIM,  DIM);  // bf16 out

  // ---- stage 2: bottleneck expansions (K=64) ----
  wmma_gemm_kernel<3, 0, 4><<<gDD, 256, 0, stream>>>(f1_bf,  Wf2_t,  gkbuf, MROWS, DIM, HDIM);   // log_sigmoid
  wmma_gemm_kernel<2, 0, 4><<<gDD, 256, 0, stream>>>(og1_bf, Wog2_t, ogbuf, MROWS, DIM, HDIM);   // sigmoid

  // ---- stage 3: per-head prep (l2 norm, exp decay, a = k*gamma*egk) ----
  prep_kernel<<<(MROWS * NHEAD) / 8, 256, 0, stream>>>(kbuf, gkbuf, gambuf, abuf);

  // ---- stage 4: sequential DPLR scan, register-resident state columns ----
  recurrence_kernel<<<BATCH * NHEAD, 64, 0, stream>>>(qbuf, kbuf, vbuf, abuf, gkbuf, obuf);

  // ---- stage 5: output gate + LayerNorm -> bf16 ----
  gate_ln_kernel<<<MROWS, 256, 0, stream>>>(obuf, ogbuf, normw, oln_bf);

  // ---- stage 6: final projection into d_out (f32) ----
  wmma_gemm_kernel<0, 0, 4><<<gDD, 256, 0, stream>>>(oln_bf, Wo_t, y, MROWS, DIM, DIM);
}
